// CustomBiGRU_6030134084276
// MI455X (gfx1250) — matrix operand order repaired and checked by reading, not compiled, not run on hardware
//
#include <hip/hip_runtime.h>

// ---------------------------------------------------------------------------
// CustomBiGRU for MI455X (gfx1250, wave32, WMMA).
//
// S=512, B=64, I=512, H=512, L=2, bidirectional.
//  * batch-parallel scan: 4 batch-slices (M=16) x 2 directions = 8 WGs/layer,
//    each WG runs the full 512-step recurrence with LDS-resident state.
//  * all GEMMs via v_wmma_f32_16x16x32_bf16 (bf16 in, f32 accum).
//  * input projection fused: A = [h | x_t], weights pre-packed as [U ; W].
//  * weights pre-swizzled to WMMA B-fragment order -> hot loop streams
//    batched contiguous 32B/lane fragments from L2 (weights ~20MB << 192MB L2).
//  * x_t tile staged into LDS via global_load_async_to_lds_b128 (ASYNCcnt),
//    issued one step ahead so the copy overlaps stage-3 compute.
// ---------------------------------------------------------------------------

typedef __bf16 bf16;
typedef __attribute__((ext_vector_type(16))) __bf16        v16bf;
typedef __attribute__((ext_vector_type(8)))  float         v8f;
typedef __attribute__((ext_vector_type(4)))  unsigned int  u32x4;

#define S_LEN 512
#define BATCH 64
#define NT    32      // N tiles in H=512
#define KB_H  16      // K blocks in H=512
#define HP    520     // LDS pitch for H-wide buffers (bf16 elems; 16B aligned, bank-skewed)

#if defined(__gfx1250__) && __has_builtin(__builtin_amdgcn_global_load_async_to_lds_b128)
#define HAS_ASYNC 1
#else
#define HAS_ASYNC 0
#endif

// The async builtin takes (AS1 int4* src, AS3 int4* dst, imm offset, imm cpol)
// (clang prints AS1 as "__device__" in HIP diagnostics).
typedef int v4i_gcc __attribute__((vector_size(16)));
typedef __attribute__((address_space(1))) v4i_gcc gv4i;
typedef __attribute__((address_space(3))) v4i_gcc lv4i;

union AFrag { v16bf v; u32x4 q[2]; };

__device__ __forceinline__ float sigm(float x) { return 1.0f / (1.0f + __expf(-x)); }

__device__ __forceinline__ void wait_async() {
#if defined(__gfx1250__)
#if __has_builtin(__builtin_amdgcn_s_wait_asynccnt)
  __builtin_amdgcn_s_wait_asynccnt(0);
#elif HAS_ASYNC
  asm volatile("s_wait_asynccnt 0x0" ::: "memory");
#endif
#endif
}

// A fragment (16x32 bf16, MxK) from an LDS activation buffer, per ISA layout:
// lanes 0-15: M=lane, K = k0..k0+7 / k0+16..k0+23 ; lanes 16-31: +8 on K.
template <int PITCH>
__device__ __forceinline__ v16bf load_a_sh(const bf16* buf, int kb, int lane) {
  const int m   = lane & 15;
  const int sel = (lane >> 4) & 1;
  const bf16* p = buf + m * PITCH + kb * 32 + sel * 8;
  AFrag a;
  a.q[0] = *(const u32x4*)(p);        // ds_load_b128
  a.q[1] = *(const u32x4*)(p + 16);   // ds_load_b128
  return a.v;
}

// B fragment (32x16 bf16, KxN), pre-swizzled: 32 contiguous bytes per lane.
// 32-bit offset so the compiler can use SGPR-base + VGPR-offset addressing.
__device__ __forceinline__ v16bf load_b(const bf16* __restrict__ mat, int kb, int nt, int lane) {
  const unsigned off = ((unsigned)(kb * NT + nt) * 32u + (unsigned)lane) << 4;
  const bf16* p = mat + off;
  AFrag b;
  b.q[0] = *(const u32x4*)(p);
  b.q[1] = *(const u32x4*)(p + 8);
  return b.v;
}

// Stage the 16 x IN_DIM bf16 x-tile for timestep t into LDS (pitch IN_DIM+8).
template <int IN_DIM>
__device__ __forceinline__ void stage_x(const bf16* __restrict__ xin, int t, int bb,
                                        bf16* xbuf, int tid) {
  constexpr int XP  = IN_DIM + 8;
  constexpr int UPR = IN_DIM / 8;       // 16B units per row
  constexpr int U_N = 16 * UPR;
#pragma unroll
  for (int u = tid; u < U_N; u += 256) {
    const int r = u / UPR;
    const int c = (u % UPR) * 8;        // element offset
    const bf16* gp = xin + ((size_t)t * BATCH + bb + r) * IN_DIM + c;
    bf16* lp = xbuf + r * XP + c;
#if HAS_ASYNC
    __builtin_amdgcn_global_load_async_to_lds_b128(
        (gv4i*)(size_t)gp, (lv4i*)(unsigned)(size_t)lp, 0, 0);
#else
    *(u32x4*)lp = *(const u32x4*)gp;
#endif
  }
}

// ---------------------------------------------------------------------------
// Recurrent layer kernel: grid = (4 batch slices, 2 directions), 256 threads.
// Dynamic LDS carve (bytes):
//   hbuf 0..16640 | abuf ..33280 | zbuf ..49920 | xbuf ..(+16*(IN_DIM+8)*2)
// ---------------------------------------------------------------------------
template <int IN_DIM>
__global__ __launch_bounds__(256)
void bigru_layer_kernel(const bf16* __restrict__ xin,
                        const bf16* __restrict__ wbase, long long cell_elems,
                        const float* __restrict__ bbase,
                        bf16* __restrict__ out_b16, float* __restrict__ out_f32)
{
  constexpr int XP   = IN_DIM + 8;
  constexpr int KB_X = IN_DIM / 32;

  extern __shared__ char smem[];
  bf16* hbuf = (bf16*)(smem);            // hidden state h
  bf16* abuf = (bf16*)(smem + 16640);    // stage-2 A (r*h), then stage-3 A (ci)
  bf16* zbuf = (bf16*)(smem + 33280);    // update gate z
  bf16* xbuf = (bf16*)(smem + 49920);    // staged x_t tile

  const int tid  = threadIdx.x;
  const int w    = tid >> 5;
  const int lane = tid & 31;
  const int dir  = blockIdx.y;
  const int bb   = blockIdx.x * 16;

  const long long Szrh = (long long)(512 + IN_DIM) * 512;
  const bf16* cw = wbase + (long long)dir * cell_elems;
  const bf16* Mz = cw;
  const bf16* Mr = cw + Szrh;
  const bf16* Mh = cw + 2 * Szrh;
  const bf16* Mg = cw + 3 * Szrh;
  const bf16* Ml = cw + 3 * Szrh + 512 * 512;

  const float* bias = bbase + dir * (5 * 512);
  const float* bz = bias;
  const float* br = bias + 512;
  const float* bh = bias + 1024;
  const float* bg = bias + 1536;
  const float* bl = bias + 2048;

  // h0 = 0, and stage x for the first step.
  for (int i = tid; i < 16 * HP; i += 256) hbuf[i] = (bf16)0.0f;
  stage_x<IN_DIM>(xin, dir ? (S_LEN - 1) : 0, bb, xbuf, tid);
  wait_async();
  __syncthreads();

  const bool isz = (w < 4);
  const int  tb1 = (w & 3) * 8;         // stage 1: 8 N-tiles per wave
  const int  tb2 = w * 4;               // stages 2/3: 4 N-tiles per wave
  const int  mlo = (lane >> 4) << 3;    // accumulator row offset (0 or 8)
  const int  nlo = lane & 15;           // accumulator column within tile
  const v8f  zero8 = {0.f, 0.f, 0.f, 0.f, 0.f, 0.f, 0.f, 0.f};

  for (int s = 0; s < S_LEN; ++s) {
    const int t = dir ? (S_LEN - 1 - s) : s;

    // ---------------- stage 1: z (waves 0-3) / r (waves 4-7), A = [h | x_t]
    const bf16* M1 = isz ? Mz : Mr;
    v8f acc[8];
#pragma unroll
    for (int i = 0; i < 8; ++i) acc[i] = zero8;

#pragma unroll 2
    for (int kb = 0; kb < KB_H; ++kb) {               // h part (LDS)
      const v16bf a = load_a_sh<HP>(hbuf, kb, lane);
      v16bf b[8];
#pragma unroll
      for (int i = 0; i < 8; ++i) b[i] = load_b(M1, kb, tb1 + i, lane);
#pragma unroll
      for (int i = 0; i < 8; ++i)
        acc[i] = __builtin_amdgcn_wmma_f32_16x16x32_bf16(false, a, false, b[i],
                                                         (short)0, acc[i], false, false);
    }
#pragma unroll 2
    for (int kx = 0; kx < KB_X; ++kx) {               // x part (LDS-staged)
      const v16bf a = load_a_sh<XP>(xbuf, kx, lane);
      v16bf b[8];
#pragma unroll
      for (int i = 0; i < 8; ++i) b[i] = load_b(M1, KB_H + kx, tb1 + i, lane);
#pragma unroll
      for (int i = 0; i < 8; ++i)
        acc[i] = __builtin_amdgcn_wmma_f32_16x16x32_bf16(false, a, false, b[i],
                                                         (short)0, acc[i], false, false);
    }
#pragma unroll
    for (int i = 0; i < 8; ++i) {
      const int n = (tb1 + i) * 16 + nlo;
      const float bv = isz ? bz[n] : br[n];
#pragma unroll
      for (int j = 0; j < 8; ++j) {
        const int m = j + mlo;
        const float v = sigm(acc[i][j] + bv);
        if (isz) zbuf[m * HP + n] = (bf16)v;
        else     abuf[m * HP + n] = (bf16)(v * (float)hbuf[m * HP + n]);   // r*h
      }
    }
    __syncthreads();

    // ---------------- stage 2: ci = [r*h | x_t] @ Mh + bh
    v8f c2[4];
#pragma unroll
    for (int i = 0; i < 4; ++i) c2[i] = zero8;
#pragma unroll 2
    for (int kb = 0; kb < KB_H; ++kb) {
      const v16bf a = load_a_sh<HP>(abuf, kb, lane);
      v16bf b[4];
#pragma unroll
      for (int i = 0; i < 4; ++i) b[i] = load_b(Mh, kb, tb2 + i, lane);
#pragma unroll
      for (int i = 0; i < 4; ++i)
        c2[i] = __builtin_amdgcn_wmma_f32_16x16x32_bf16(false, a, false, b[i],
                                                        (short)0, c2[i], false, false);
    }
#pragma unroll 2
    for (int kx = 0; kx < KB_X; ++kx) {
      const v16bf a = load_a_sh<XP>(xbuf, kx, lane);
      v16bf b[4];
#pragma unroll
      for (int i = 0; i < 4; ++i) b[i] = load_b(Mh, KB_H + kx, tb2 + i, lane);
#pragma unroll
      for (int i = 0; i < 4; ++i)
        c2[i] = __builtin_amdgcn_wmma_f32_16x16x32_bf16(false, a, false, b[i],
                                                        (short)0, c2[i], false, false);
    }
    __syncthreads();   // all reads of r*h and x_t for this step are done

    // xbuf is now dead for this step: prefetch next step's x tile (async DMA
    // overlaps the ci write + stage-3 GEMM below; completion enforced at the
    // end-of-step wait_async + barrier).
    if (s + 1 < S_LEN) stage_x<IN_DIM>(xin, dir ? (t - 1) : (t + 1), bb, xbuf, tid);

#pragma unroll
    for (int i = 0; i < 4; ++i) {
      const int n = (tb2 + i) * 16 + nlo;
      const float bv = bh[n];
#pragma unroll
      for (int j = 0; j < 8; ++j) {
        const int m = j + mlo;
        abuf[m * HP + n] = (bf16)(c2[i][j] + bv);                          // ci
      }
    }
    __syncthreads();

    // ---------------- stage 3: gate = sigmoid(ci@g), cand = (ci@l)*gate, combine
    v8f cg[4], cl[4];
#pragma unroll
    for (int i = 0; i < 4; ++i) { cg[i] = zero8; cl[i] = zero8; }
#pragma unroll 2
    for (int kb = 0; kb < KB_H; ++kb) {
      const v16bf a = load_a_sh<HP>(abuf, kb, lane);
      v16bf b1[4], b2[4];
#pragma unroll
      for (int i = 0; i < 4; ++i) { b1[i] = load_b(Mg, kb, tb2 + i, lane);
                                    b2[i] = load_b(Ml, kb, tb2 + i, lane); }
#pragma unroll
      for (int i = 0; i < 4; ++i) {
        cg[i] = __builtin_amdgcn_wmma_f32_16x16x32_bf16(false, a, false, b1[i],
                                                        (short)0, cg[i], false, false);
        cl[i] = __builtin_amdgcn_wmma_f32_16x16x32_bf16(false, a, false, b2[i],
                                                        (short)0, cl[i], false, false);
      }
    }
#pragma unroll
    for (int i = 0; i < 4; ++i) {
      const int n = (tb2 + i) * 16 + nlo;
      const float bgv = bg[n], blv = bl[n];
#pragma unroll
      for (int j = 0; j < 8; ++j) {
        const int m = j + mlo;
        const float gate = sigm(cg[i][j] + bgv);
        const float cand = (cl[i][j] + blv) * gate;
        const float zv = (float)zbuf[m * HP + n];
        const float hv = (float)hbuf[m * HP + n];
        const float hn = zv * cand + (1.0f - zv) * hv;
        hbuf[m * HP + n] = (bf16)hn;
        const size_t o = ((size_t)t * BATCH + bb + m) * 1024 + (size_t)dir * 512 + n;
        if (out_b16) out_b16[o] = (bf16)hn;
        if (out_f32) out_f32[o] = hn;
      }
    }
    wait_async();      // own async x-loads done before the barrier...
    __syncthreads();   // ...so after it, every wave sees the full next x tile + h
  }
}

// ---------------------------------------------------------------------------
// One-shot conversion kernels
// ---------------------------------------------------------------------------

// Pack combined [U ; W] (ktot x 512, fp32 row-major) into WMMA B-fragment
// order: frag(kb,nt) = 32 lanes x 16 bf16; lanes 0-15 hold K=kb*32+{0..7,16..23},
// lanes 16-31 hold K=kb*32+{8..15,24..31}, N = nt*16 + (lane&15).
__global__ void swizzle_weights_kernel(const float* __restrict__ U,
                                       const float* __restrict__ W,
                                       bf16* __restrict__ dst, int ktot)
{
  const int tid = blockIdx.x * blockDim.x + threadIdx.x;
  const int total = (ktot >> 5) * 8192;
  if (tid >= total) return;
  const int d    = tid & 7;
  const int lane = (tid >> 3) & 31;
  const int nt   = (tid >> 8) & 31;
  const int kb   = tid >> 13;
  const int n    = nt * 16 + (lane & 15);
  const int k    = kb * 32 + (d >> 2) * 16 + ((lane >> 4) & 1) * 8 + (d & 3) * 2;
  float f0, f1;
  if (k < 512) { f0 = U[(size_t)k * 512 + n];         f1 = U[(size_t)(k + 1) * 512 + n]; }
  else         { f0 = W[(size_t)(k - 512) * 512 + n]; f1 = W[(size_t)(k - 511) * 512 + n]; }
  const size_t o = ((((size_t)kb * 32 + nt) * 32 + lane) << 4) + d * 2;
  dst[o]     = (bf16)f0;
  dst[o + 1] = (bf16)f1;
}

// Pre-combined biases per cell: [Wz_b+Uz_b | Wr_b+Ur_b | Wh_b+Uh_b+bias_h | g_b | l_b]
__global__ void bias_combine_kernel(const float* wzb, const float* uzb,
                                    const float* wrb, const float* urb,
                                    const float* whb, const float* uhb, const float* bh2,
                                    const float* gb,  const float* lb,
                                    float* __restrict__ out)
{
  const int i = blockIdx.x * blockDim.x + threadIdx.x;
  if (i >= 512) return;
  out[i]        = wzb[i] + uzb[i];
  out[512 + i]  = wrb[i] + urb[i];
  out[1024 + i] = whb[i] + uhb[i] + bh2[i];
  out[1536 + i] = gb[i];
  out[2048 + i] = lb[i];
}

__global__ void f32_to_bf16_kernel(const float* __restrict__ in,
                                   bf16* __restrict__ out, int n)
{
  const int i = blockIdx.x * blockDim.x + threadIdx.x;
  if (i < n) out[i] = (bf16)in[i];
}

// ---------------------------------------------------------------------------
// Host launcher
// ---------------------------------------------------------------------------
extern "C" void kernel_launch(void* const* d_in, const int* in_sizes, int n_in,
                              void* d_out, int out_size, void* d_ws, size_t ws_size,
                              hipStream_t stream)
{
  (void)in_sizes; (void)n_in; (void)out_size; (void)ws_size;

  const float* x = (const float*)d_in[0];
  // cell order: L0.fwd, L0.bwd, L1.fwd, L1.bwd; 17 arrays per cell:
  // 0 Wz_w 1 Wz_b 2 Uz_w 3 Uz_b 4 Wr_w 5 Wr_b 6 Ur_w 7 Ur_b
  // 8 Wh_w 9 Wh_b 10 Uh_w 11 Uh_b 12 g_w 13 g_b 14 l_w 15 l_b 16 bias_h
  auto P = [&](int cell, int j) -> const float* {
    return (const float*)d_in[1 + cell * 17 + j];
  };

  char* ws = (char*)d_ws;
  size_t off = 0;
  bf16* x0 = (bf16*)(ws + off); off += (size_t)512 * 64 * 512 * 2;    // layer-0 input, bf16
  bf16* x1 = (bf16*)(ws + off); off += (size_t)512 * 64 * 1024 * 2;   // layer-1 input, bf16
  const long long cellE0 = 3LL * 1024 * 512 + 2LL * 512 * 512;        // elems per L0 cell
  const long long cellE1 = 3LL * 1536 * 512 + 2LL * 512 * 512;        // elems per L1 cell
  bf16* W0 = (bf16*)(ws + off); off += (size_t)cellE0 * 2 * 2;
  bf16* W1 = (bf16*)(ws + off); off += (size_t)cellE1 * 2 * 2;
  float* Bs = (float*)(ws + off); off += (size_t)4 * 5 * 512 * 4;

  // x -> bf16
  {
    const int n = 512 * 64 * 512;
    f32_to_bf16_kernel<<<(n + 255) / 256, 256, 0, stream>>>(x, x0, n);
  }

  // weight pre-swizzle + bias combine (one shot, tiny vs. the scan)
  for (int cell = 0; cell < 4; ++cell) {
    const int layer  = cell >> 1;
    const int in_dim = layer ? 1024 : 512;
    const int ktot   = 512 + in_dim;
    const long long ce   = layer ? cellE1 : cellE0;
    const long long Szrh = (long long)ktot * 512;
    bf16* base = (layer ? W1 : W0) + (cell & 1) * ce;

    struct { const float* U; const float* W; bf16* dst; int kt; } mm[5] = {
      { P(cell, 2),  P(cell, 0), base,                        ktot },  // [Uz;Wz]
      { P(cell, 6),  P(cell, 4), base + Szrh,                 ktot },  // [Ur;Wr]
      { P(cell, 10), P(cell, 8), base + 2 * Szrh,             ktot },  // [Uh;Wh]
      { P(cell, 12), nullptr,    base + 3 * Szrh,             512  },  // g
      { P(cell, 14), nullptr,    base + 3 * Szrh + 512 * 512, 512  },  // l
    };
    for (int m = 0; m < 5; ++m) {
      const int total = (mm[m].kt >> 5) * 8192;
      swizzle_weights_kernel<<<(total + 255) / 256, 256, 0, stream>>>(
          mm[m].U, mm[m].W, mm[m].dst, mm[m].kt);
    }
    bias_combine_kernel<<<2, 256, 0, stream>>>(
        P(cell, 1), P(cell, 3), P(cell, 5), P(cell, 7),
        P(cell, 9), P(cell, 11), P(cell, 16),
        P(cell, 13), P(cell, 15), Bs + cell * 5 * 512);
  }

  // Dynamic LDS: 49920 state bytes + 16*(IN_DIM+8)*2 x-stage bytes.
  const int smem0 = 49920 + 16 * (512 + 8) * 2;     // 66560
  const int smem1 = 49920 + 16 * (1024 + 8) * 2;    // 82944

  // layer 0: bf16 output becomes layer-1 input (fwd cols 0..511, bwd 512..1023)
  bigru_layer_kernel<512><<<dim3(4, 2), 256, smem0, stream>>>(
      x0, W0, cellE0, Bs, x1, nullptr);
  // layer 1: fp32 output -> d_out (S,B,1024)
  bigru_layer_kernel<1024><<<dim3(4, 2), 256, smem1, stream>>>(
      x1, W1, cellE1, Bs + 2 * 5 * 512, nullptr, (float*)d_out);
}
